// SAGE_120259084569
// MI455X (gfx1250) — matrix operand order repaired and checked by reading, not compiled, not run on hardware
//
#include <hip/hip_runtime.h>
#include <hip/hip_bf16.h>

typedef __attribute__((ext_vector_type(16))) _Float16 v16h;
typedef __attribute__((ext_vector_type(8)))  float    v8f;

#define FEAT 128   // feature width entering both aggregation passes

// ---------------------------------------------------------------------------
// Zero-fill a float buffer (scratch agg / cnt must start at 0 every call).
// ---------------------------------------------------------------------------
__global__ void sage_zero_f32(float* __restrict__ p, long long n) {
  long long i = (long long)blockIdx.x * blockDim.x + threadIdx.x;
  long long stride = (long long)gridDim.x * blockDim.x;
  for (; i < n; i += stride) p[i] = 0.0f;
}

// ---------------------------------------------------------------------------
// Edge scatter: one wave32 per edge. Each lane handles 4 consecutive floats
// (float4 gather, 4x global_atomic_add_f32). Lane 0 optionally counts degree.
// ---------------------------------------------------------------------------
__global__ void sage_scatter(const long long* __restrict__ src,
                             const long long* __restrict__ dst,
                             const float* __restrict__ feat,
                             float* __restrict__ agg,
                             float* __restrict__ cnt,
                             int nEdges, int doCnt) {
  int tid    = blockIdx.x * blockDim.x + threadIdx.x;
  int lane   = tid & 31;
  long long e = tid >> 5;
  long long nWaves = ((long long)gridDim.x * blockDim.x) >> 5;
  for (; e < nEdges; e += nWaves) {
    long long s = src[e];
    long long d = dst[e];
    float4 v = *((const float4*)(feat + s * FEAT) + lane);
    float* ap = agg + d * FEAT + lane * 4;
    __hip_atomic_fetch_add(ap + 0, v.x, __ATOMIC_RELAXED, __HIP_MEMORY_SCOPE_AGENT);
    __hip_atomic_fetch_add(ap + 1, v.y, __ATOMIC_RELAXED, __HIP_MEMORY_SCOPE_AGENT);
    __hip_atomic_fetch_add(ap + 2, v.z, __ATOMIC_RELAXED, __HIP_MEMORY_SCOPE_AGENT);
    __hip_atomic_fetch_add(ap + 3, v.w, __ATOMIC_RELAXED, __HIP_MEMORY_SCOPE_AGENT);
    if (doCnt && lane == 0)
      __hip_atomic_fetch_add(cnt + d, 1.0f, __ATOMIC_RELAXED, __HIP_MEMORY_SCOPE_AGENT);
  }
}

// ---------------------------------------------------------------------------
// invc[i] = 1 / max(cnt[i], 1)
// ---------------------------------------------------------------------------
__global__ void sage_invcnt(const float* __restrict__ cnt,
                            float* __restrict__ invc, int n) {
  int i = blockIdx.x * blockDim.x + threadIdx.x;
  if (i < n) invc[i] = 1.0f / fmaxf(cnt[i], 1.0f);
}

// ---------------------------------------------------------------------------
// Pre-swizzle a row-major f32 weight W[K][N] into f16 WMMA-B panels:
//   P[((nt*(K/32)+kt)*32 + lane)*16 + j] = W[kt*32 + (j>>3)*16 + (lane>>4)*8 + (j&7)][nt*16 + (lane&15)]
// so each wave lane's 16-half B fragment is one contiguous 32-byte read.
// ---------------------------------------------------------------------------
__global__ void sage_swizzleB(const float* __restrict__ W,
                              _Float16* __restrict__ P, int K, int N) {
  int idx = blockIdx.x * blockDim.x + threadIdx.x;
  if (idx >= K * N) return;
  int j    = idx & 15;
  int lane = (idx >> 4) & 31;
  int rest = idx >> 9;                 // nt*(K/32) + kt
  int kt   = rest % (K / 32);
  int nt   = rest / (K / 32);
  int k = kt * 32 + (j >> 3) * 16 + (lane >> 4) * 8 + (j & 7);
  int n = nt * 16 + (lane & 15);
  P[idx] = (_Float16)W[(long long)k * N + n];
}

// ---------------------------------------------------------------------------
// Fused SAGE layer:  out = act( (agg .* invc) @ Wl + bl + x @ Wr )
// Block covers M=32 rows; NDIM/16 waves. Each wave owns one 16-col N-tile and
// TWO 16-row accumulators, so every B fragment (one v16h = 2x b128 load from
// the pre-swizzled panel) feeds two v_wmma_f32_16x16x32_f16 ops.
// ---------------------------------------------------------------------------
template <int KDIM, int NDIM, bool RELU>
__global__ void sage_layer(const float* __restrict__ agg,
                           const float* __restrict__ invc,
                           const float* __restrict__ xin,
                           const _Float16* __restrict__ Pl,   // swizzled Wl
                           const float* __restrict__ bl,
                           const _Float16* __restrict__ Pr,   // swizzled Wr
                           float* __restrict__ out, int nNodes) {
  constexpr int KP = KDIM + 8;               // +16B row skew vs 64-bank LDS
  constexpr int KT = KDIM / 32;              // K-steps
  __shared__ _Float16 sMean[32][KP];
  __shared__ _Float16 sX[32][KP];

  const int tid   = threadIdx.x;
  const int mbase = blockIdx.x * 32;

  // Stage A operands (mean-aggregated neighbors + root features) as f16.
  for (int i = tid; i < 32 * KDIM; i += blockDim.x) {
    int row = i / KDIM, k = i % KDIM;
    int node = mbase + row;
    float m = 0.0f, xv = 0.0f;
    if (node < nNodes) {
      float ic = invc[node];
      m  = agg[(long long)node * KDIM + k] * ic;
      xv = xin[(long long)node * KDIM + k];
    }
    sMean[row][k] = (_Float16)m;
    sX[row][k]    = (_Float16)xv;
  }
  __syncthreads();

  const int lane  = tid & 31;
  const int ntile = tid >> 5;                // one wave per 16-col tile
  const int col   = ntile * 16 + (lane & 15);
  const int row   = lane & 15;
  const int ksel  = lane >> 4;               // lane half-select (ISA 7.12.2)

  float bias = bl[col];
  v8f acc0, acc1;                            // rows mbase+0..15 / +16..31
#pragma unroll
  for (int r = 0; r < 8; ++r) { acc0[r] = bias; acc1[r] = bias; }

#pragma unroll
  for (int kti = 0; kti < KT; ++kti) {
    const int kt = kti * 32;
    // A 16x32 f16 fragments (ISA layout): halves 0-7 -> K kt+ksel*8..,
    // halves 8-15 -> K kt+16+ksel*8.. ; two M-tiles.
    v16h aM0{}, aX0{}, aM1{}, aX1{};
#pragma unroll
    for (int j = 0; j < 8; ++j) {
      int kA = kt + ksel * 8 + j;
      int kB = kt + 16 + ksel * 8 + j;
      aM0[j]     = sMean[row][kA];      aM0[8 + j] = sMean[row][kB];
      aX0[j]     = sX[row][kA];         aX0[8 + j] = sX[row][kB];
      aM1[j]     = sMean[row + 16][kA]; aM1[8 + j] = sMean[row + 16][kB];
      aX1[j]     = sX[row + 16][kA];    aX1[8 + j] = sX[row + 16][kB];
    }
    // B fragments: contiguous 32B per lane from the swizzled panel.
    const long long pbase = ((long long)(ntile * KT + kti) * 32 + lane) * 16;
    v16h bL = *(const v16h*)(Pl + pbase);
    v16h bR = *(const v16h*)(Pr + pbase);

    acc0 = __builtin_amdgcn_wmma_f32_16x16x32_f16(false, aM0, false, bL,
                                                  (short)0, acc0, false, false);
    acc0 = __builtin_amdgcn_wmma_f32_16x16x32_f16(false, aX0, false, bR,
                                                  (short)0, acc0, false, false);
    acc1 = __builtin_amdgcn_wmma_f32_16x16x32_f16(false, aM1, false, bL,
                                                  (short)0, acc1, false, false);
    acc1 = __builtin_amdgcn_wmma_f32_16x16x32_f16(false, aX1, false, bR,
                                                  (short)0, acc1, false, false);
  }

  // D 16x16 f32: VGPR r -> M=r (lanes 0-15) / M=8+r (lanes 16-31).
#pragma unroll
  for (int r = 0; r < 8; ++r) {
    int m = (lane < 16) ? r : (8 + r);
    float v0 = acc0[r], v1 = acc1[r];
    if (RELU) { v0 = fmaxf(v0, 0.0f); v1 = fmaxf(v1, 0.0f); }
    int n0 = mbase + m, n1 = mbase + 16 + m;
    if (n0 < nNodes) out[(long long)n0 * NDIM + col] = v0;
    if (n1 < nNodes) out[(long long)n1 * NDIM + col] = v1;
  }
}

// ---------------------------------------------------------------------------
extern "C" void kernel_launch(void* const* d_in, const int* in_sizes, int n_in,
                              void* d_out, int out_size, void* d_ws, size_t ws_size,
                              hipStream_t stream) {
  const float*     x   = (const float*)d_in[0];
  const long long* ei  = (const long long*)d_in[1];   // int64 (2, E)
  const float*     Wl0 = (const float*)d_in[2];
  const float*     bl0 = (const float*)d_in[3];
  const float*     Wr0 = (const float*)d_in[4];
  const float*     Wl1 = (const float*)d_in[5];
  const float*     bl1 = (const float*)d_in[6];
  const float*     Wr1 = (const float*)d_in[7];

  const int N = in_sizes[0] / FEAT;     // 100000
  const int E = in_sizes[1] / 2;        // 1600000
  const long long NF = (long long)N * FEAT;

  float* agg  = (float*)d_ws;           // N*128 f32
  float* h    = agg + NF;               // N*128 f32
  float* cnt  = h + NF;                 // N f32
  float* invc = cnt + N;                // N f32
  _Float16* Pl0 = (_Float16*)(invc + N);        // 128x128 f16 panels
  _Float16* Pr0 = Pl0 + 128 * 128;
  _Float16* Pl1 = Pr0 + 128 * 128;              // 128x64
  _Float16* Pr1 = Pl1 + 128 * 64;

  const long long* srcIdx = ei;
  const long long* dstIdx = ei + E;

  const int ZB = 256;
  int zeroBlocksAgg = (int)((NF + ZB - 1) / ZB);
  int zeroBlocksCnt = (N + ZB - 1) / ZB;
  int scatterBlocks = (E + 7) / 8;      // 8 waves (edges) per 256-thread block
  int mtiles        = (N + 31) / 32;    // 32 rows per block

  // Weight pre-swizzle into WMMA-B f16 panels (tiny, L2-resident).
  sage_swizzleB<<<(128 * 128 + 255) / 256, 256, 0, stream>>>(Wl0, Pl0, 128, 128);
  sage_swizzleB<<<(128 * 128 + 255) / 256, 256, 0, stream>>>(Wr0, Pr0, 128, 128);
  sage_swizzleB<<<(128 * 64 + 255) / 256, 256, 0, stream>>>(Wl1, Pl1, 128, 64);
  sage_swizzleB<<<(128 * 64 + 255) / 256, 256, 0, stream>>>(Wr1, Pr1, 128, 64);

  // ---- Layer 0 ----
  sage_zero_f32<<<zeroBlocksAgg, ZB, 0, stream>>>(agg, NF);
  sage_zero_f32<<<zeroBlocksCnt, ZB, 0, stream>>>(cnt, N);
  sage_scatter<<<scatterBlocks, 256, 0, stream>>>(srcIdx, dstIdx, x, agg, cnt, E, 1);
  sage_invcnt<<<zeroBlocksCnt, ZB, 0, stream>>>(cnt, invc, N);
  sage_layer<128, 128, true><<<mtiles, 256, 0, stream>>>(
      agg, invc, x, Pl0, bl0, Pr0, h, N);

  // ---- Layer 1 ----
  sage_zero_f32<<<zeroBlocksAgg, ZB, 0, stream>>>(agg, NF);
  sage_scatter<<<scatterBlocks, 256, 0, stream>>>(srcIdx, dstIdx, h, agg, cnt, E, 0);
  sage_layer<128, 64, false><<<mtiles, 128, 0, stream>>>(
      agg, invc, h, Pl1, bl1, Pr1, (float*)d_out, N);
}
